// Generator_45466523796089
// MI455X (gfx1250) — compile-verified, matrix operand order checked by reading, not run on hardware
//
#include <hip/hip_runtime.h>
#include <cmath>

// ---------------------------------------------------------------------------
// Social-LSTM trajectory generator for MI455X (gfx1250, wave32, WMMA).
//
// Heavy GEMMs -> v_wmma_f32_16x16x32_f16 (fp32 accumulate), 64x16 macro-tile
// per wave (4 accumulators, weight-fragment reuse), compile-time K *and*
// strides so every fragment load is a global_load_b128 with an immediate
// offset off one per-lane base pointer (no per-iteration 64-bit address
// math).  Everything else (gates, softmax, K=4 / K=3 skinny layers) ->
// scalar/wave VALU kernels.  The O(N^2) social MLP is restricted to the
// block-diagonal (within-group) pairs the reference actually uses, fully
// fused in registers so the 25MB pairwise feature tensor never exists.
// ---------------------------------------------------------------------------

typedef __attribute__((ext_vector_type(16))) _Float16 v16h;
typedef __attribute__((ext_vector_type(8)))  float    v8f;

#define NAGENT 768
#define TOBS   8
#define HDIM   128
#define FDIM   128
#define NZDIM  64
#define GSIZE  64
#define NGROUP 12          // NAGENT / GSIZE
#define DDIM   320         // HDIM + FDIM + NZDIM
#define NNEXT  12
#define MBLK   64          // rows per wave (4 x 16 sub-tiles)

__device__ __forceinline__ float sigmoidf(float x) { return 1.0f / (1.0f + __expf(-x)); }

// Convert a 16x32 fragment (two contiguous 8-float runs at p[0..7], p[16..23])
// from f32 memory to the ISA 7.12.2 16-bit A/B packing, per lane:
//   lanes 0-15  : row = lane,    K = {k0+0..7,  k0+16..23}
//   lanes 16-31 : row = lane-16, K = {k0+8..15, k0+24..31}
// (the +8 half-lane shift is baked into the per-lane base pointer).
// B (32x16, from the row-major NxK weight matrix) uses the symmetric packing;
// layout assumption flagged for hardware validation (compile-only loop).
__device__ __forceinline__ v16h frag_from(const float* __restrict__ p)
{
    float4 c0 = ((const float4*)p)[0];
    float4 c1 = ((const float4*)p)[1];
    float4 c2 = ((const float4*)(p + 16))[0];
    float4 c3 = ((const float4*)(p + 16))[1];
    v16h v;
    v[0]  = (_Float16)c0.x; v[1]  = (_Float16)c0.y; v[2]  = (_Float16)c0.z; v[3]  = (_Float16)c0.w;
    v[4]  = (_Float16)c1.x; v[5]  = (_Float16)c1.y; v[6]  = (_Float16)c1.z; v[7]  = (_Float16)c1.w;
    v[8]  = (_Float16)c2.x; v[9]  = (_Float16)c2.y; v[10] = (_Float16)c2.z; v[11] = (_Float16)c2.w;
    v[12] = (_Float16)c3.x; v[13] = (_Float16)c3.y; v[14] = (_Float16)c3.z; v[15] = (_Float16)c3.w;
    return v;
}

// Z[768 x N] = act( X1 @ W1^T (+ X2 @ W2^T) + b1 (+ b2) )
// W row-major (N x K).  K and X strides are template constants: all fragment
// addresses are immediate offsets from one per-lane base.  One wave per
// 64x16 output strip; B fragment reused by 4 WMMAs.  act: 0 none, 1 relu,
// 2 leaky-relu(0.2).
template <int K1, int LDX1, int K2, int LDX2>
__global__ void __launch_bounds__(256)
gemm_wmma_kernel(const float* __restrict__ X1, const float* __restrict__ W1,
                 const float* __restrict__ X2, const float* __restrict__ W2,
                 const float* __restrict__ b1, const float* __restrict__ b2,
                 float* __restrict__ Z, int ldz,
                 int Ntiles, int act)
{
    int wave = (int)((blockIdx.x * blockDim.x + threadIdx.x) >> 5);
    int lane = (int)(threadIdx.x & 31);
    if (wave >= (NAGENT / MBLK) * Ntiles) return;   // wave-uniform: EXEC all-1s
    int mb = wave / Ntiles, tn = wave % Ntiles;
    int m0 = mb * MBLK, n0 = tn * 16;

    int rsel = lane & 15;
    int ksel = (lane >> 4) << 3;                     // 0 or 8 (half-lane K shift)
    const float* baseA1 = X1 + (size_t)(m0 + rsel) * LDX1 + ksel;
    const float* baseB1 = W1 + (size_t)(n0 + rsel) * K1 + ksel;

    v8f acc[4] = {{}, {}, {}, {}};
#pragma unroll
    for (int k = 0; k < K1; k += 32) {
        if (k + 32 < K1)   // pull next weight chunk toward L2/L0
            __builtin_prefetch(baseB1 + k + 32, 0, 1);
        v16h b = frag_from(baseB1 + k);
#pragma unroll
        for (int s = 0; s < 4; ++s) {
            v16h a = frag_from(baseA1 + 16 * s * LDX1 + k);   // imm offsets
            acc[s] = __builtin_amdgcn_wmma_f32_16x16x32_f16(false, a, false, b,
                                                            (short)0, acc[s], false, false);
        }
    }
    if constexpr (K2 > 0) {   // fused second accumulation (x@W_ih^T + h@W_hh^T)
        const float* baseA2 = X2 + (size_t)(m0 + rsel) * LDX2 + ksel;
        const float* baseB2 = W2 + (size_t)(n0 + rsel) * K2 + ksel;
#pragma unroll
        for (int k = 0; k < K2; k += 32) {
            v16h b = frag_from(baseB2 + k);
#pragma unroll
            for (int s = 0; s < 4; ++s) {
                v16h a = frag_from(baseA2 + 16 * s * LDX2 + k);
                acc[s] = __builtin_amdgcn_wmma_f32_16x16x32_f16(false, a, false, b,
                                                                (short)0, acc[s], false, false);
            }
        }
    }

    // C/D layout: lanes 0-15 -> N=lane, M=+r ; lanes 16-31 -> N=lane-16, M=+8+r
    int n = n0 + rsel;
    float bias = 0.0f;
    if (b1) bias += b1[n];
    if (b2) bias += b2[n];
#pragma unroll
    for (int s = 0; s < 4; ++s) {
        int mbase = m0 + 16 * s + ksel;
#pragma unroll
        for (int r = 0; r < 8; ++r) {
            float v = acc[s][r] + bias;
            if (act == 1)      v = fmaxf(v, 0.0f);
            else if (act == 2) v = (v >= 0.0f) ? v : 0.2f * v;
            Z[(size_t)(mbase + r) * ldz + n] = v;
        }
    }
}

#define GEMM(K1, LDX1, K2, LDX2, X1, W1, X2, W2, B1, B2, ZP, LDZ, NN, ACT)          \
    do {                                                                            \
        int nt_ = (NN) / 16;                                                        \
        int blocks_ = ((NAGENT / MBLK) * nt_ * 32 + 255) / 256;                     \
        gemm_wmma_kernel<K1, LDX1, K2, LDX2><<<blocks_, 256, 0, stream>>>(          \
            X1, W1, X2, W2, B1, B2, ZP, LDZ, nt_, ACT);                             \
    } while (0)

// out[M x 128] = in4[M x 4] @ W_emb^T + b_emb   (K=4: scalar)
__global__ void embed_kernel(const float* __restrict__ in4,
                             const float* __restrict__ Wemb,
                             const float* __restrict__ bemb,
                             float* __restrict__ out, int M)
{
    int idx = blockIdx.x * blockDim.x + threadIdx.x;
    if (idx >= M * HDIM) return;
    int m = idx / HDIM, hc = idx % HDIM;
    const float* xr = in4 + (size_t)m * 4;
    const float* wr = Wemb + (size_t)hc * 4;
    out[idx] = bemb[hc] + xr[0]*wr[0] + xr[1]*wr[1] + xr[2]*wr[2] + xr[3]*wr[3];
}

// LSTM pointwise gates from z[M x 512]; h,c updated in place (next launch).
__global__ void lstm_gates_kernel(const float* __restrict__ z,
                                  float* __restrict__ h, float* __restrict__ c)
{
    int idx = blockIdx.x * blockDim.x + threadIdx.x;
    if (idx >= NAGENT * HDIM) return;
    int m = idx / HDIM, j = idx % HDIM;
    const float* zr = z + (size_t)m * 4 * HDIM;
    float zi = zr[j], zf = zr[j + HDIM], zg = zr[j + 2*HDIM], zo = zr[j + 3*HDIM];
    float cn = sigmoidf(zf) * c[idx] + sigmoidf(zi) * tanhf(zg);
    c[idx] = cn;
    h[idx] = sigmoidf(zo) * tanhf(cn);
}

// h0=c0=0, last = obsv[:, T-1, :]
__global__ void init_kernel(const float* __restrict__ obsv,
                            float* __restrict__ h, float* __restrict__ c,
                            float* __restrict__ last)
{
    int idx = blockIdx.x * blockDim.x + threadIdx.x;
    if (idx < NAGENT * HDIM) { h[idx] = 0.0f; c[idx] = 0.0f; }
    if (idx < NAGENT * 4)
        last[idx] = obsv[(size_t)(idx >> 2) * TOBS * 4 + (TOBS - 1) * 4 + (idx & 3)];
}

// Fused: pairwise social feats (l2,cos,dca) -> 3->32->64->128 MLP -> dot with
// Wh[j] = (h@Wa^T+ba)[j].  Only within-group pairs (block diagonal of the
// reference's N^2 computation).  Diagonal masked to -1000 as in reference.
__global__ void social_sigma_kernel(const float* __restrict__ obsv,
                                    const float* __restrict__ Wf1, const float* __restrict__ bf1,
                                    const float* __restrict__ Wf2, const float* __restrict__ bf2,
                                    const float* __restrict__ Wf3, const float* __restrict__ bf3,
                                    const float* __restrict__ Wh,
                                    float* __restrict__ sigma)
{
    int idx = blockIdx.x * blockDim.x + threadIdx.x;   // b*G*G + i*G + j
    if (idx >= NGROUP * GSIZE * GSIZE) return;
    int j = idx % GSIZE, i = (idx / GSIZE) % GSIZE, b = idx / (GSIZE * GSIZE);
    if (i == j) { sigma[idx] = -1000.0f; return; }
    int gi = b * GSIZE + i, gj = b * GSIZE + j;
    const float* xi = obsv + (size_t)gi * TOBS * 4 + (TOBS - 1) * 4;
    const float* xj = obsv + (size_t)gj * TOBS * 4 + (TOBS - 1) * 4;
    float dpx = xi[0] - xj[0], dpy = xi[1] - xj[1];
    float dvx = xi[2] - xj[2], dvy = xi[3] - xj[3];
    float l2  = sqrtf(dpx*dpx + dpy*dpy);
    float vx  = xi[2], vy = xi[3];
    float vn  = sqrtf(vx*vx + vy*vy);
    float cosv = (dpx*vx + dpy*vy) / (l2 * vn + 1e-6f);
    float ttca = -(dpx*dvx + dpy*dvy) / (dvx*dvx + dvy*dvy + 1e-6f);
    float cax = dpx + ttca*dvx, cay = dpy + ttca*dvy;
    float dca = sqrtf(cax*cax + cay*cay);

    float e1[32];
#pragma unroll
    for (int a = 0; a < 32; ++a) {
        float s = bf1[a] + Wf1[a*3+0]*l2 + Wf1[a*3+1]*cosv + Wf1[a*3+2]*dca;
        e1[a] = fmaxf(s, 0.0f);
    }
    float e2[64];
    for (int a = 0; a < 64; ++a) {
        float s = bf2[a];
        const float* w = Wf2 + a * 32;
#pragma unroll
        for (int q = 0; q < 32; ++q) s += w[q] * e1[q];
        e2[a] = fmaxf(s, 0.0f);
    }
    float sg = 0.0f;
    const float* whr = Wh + (size_t)gj * FDIM;
    for (int a = 0; a < FDIM; ++a) {
        float s = bf3[a];
        const float* w = Wf3 + a * 64;
#pragma unroll
        for (int q = 0; q < 64; ++q) s += w[q] * e2[q];
        sg += s * whr[a];
    }
    sigma[idx] = sg;
}

// Wave-parallel softmax over G=64 per row (in place): one wave per (b,i) row,
// 2 elements per lane, __shfl_xor butterfly reductions (lane-permute path).
__global__ void __launch_bounds__(256) softmax_rows_kernel(float* __restrict__ sigma)
{
    int wave = (int)((blockIdx.x * blockDim.x + threadIdx.x) >> 5);
    int lane = (int)(threadIdx.x & 31);
    if (wave >= NGROUP * GSIZE) return;
    float* r = sigma + (size_t)wave * GSIZE;
    float a = r[lane], b = r[lane + 32];
    float mx = fmaxf(a, b);
#pragma unroll
    for (int off = 16; off > 0; off >>= 1) mx = fmaxf(mx, __shfl_xor(mx, off, 32));
    float ea = __expf(a - mx), eb = __expf(b - mx);
    float sum = ea + eb;
#pragma unroll
    for (int off = 16; off > 0; off >>= 1) sum += __shfl_xor(sum, off, 32);
    float inv = 1.0f / sum;
    r[lane] = ea * inv;
    r[lane + 32] = eb * inv;
}

// weighted[gi, col] = sum_j attn[b,i,j] * h[b*G+j, col]
__global__ void attn_apply_kernel(const float* __restrict__ attn,
                                  const float* __restrict__ h,
                                  float* __restrict__ weighted)
{
    int idx = blockIdx.x * blockDim.x + threadIdx.x;
    if (idx >= NAGENT * HDIM) return;
    int gi = idx / HDIM, col = idx % HDIM;
    int b = gi / GSIZE, i = gi % GSIZE;
    const float* ar = attn + ((size_t)b * GSIZE + i) * GSIZE;
    const float* hb = h + (size_t)b * GSIZE * HDIM + col;
    float s = 0.0f;
    for (int q = 0; q < GSIZE; ++q) s += ar[q] * hb[(size_t)q * HDIM];
    weighted[idx] = s;
}

// u0[:, 128:320] = [weighted | noise]   (fixed across decoder steps)
__global__ void u0_static_kernel(const float* __restrict__ weighted,
                                 const float* __restrict__ noise,
                                 float* __restrict__ u0)
{
    int idx = blockIdx.x * blockDim.x + threadIdx.x;
    if (idx >= NAGENT * (FDIM + NZDIM)) return;
    int m = idx / (FDIM + NZDIM), col = idx % (FDIM + NZDIM);
    u0[(size_t)m * DDIM + HDIM + col] =
        (col < FDIM) ? weighted[(size_t)m * FDIM + col]
                     : noise[(size_t)m * NZDIM + (col - FDIM)];
}

// u0[:, 0:128] = h   (per decoder step)
__global__ void u0_h_kernel(const float* __restrict__ h, float* __restrict__ u0)
{
    int idx = blockIdx.x * blockDim.x + threadIdx.x;
    if (idx >= NAGENT * HDIM) return;
    int m = idx / HDIM, col = idx % HDIM;
    u0[(size_t)m * DDIM + col] = h[idx];
}

// v = d3 @ Wd4^T + bd4 (N=2); p = v + last[:, :2]; new_last=[p,v]; emit preds
__global__ void dec_tail_kernel(const float* __restrict__ d3,
                                const float* __restrict__ Wd4,
                                const float* __restrict__ bd4,
                                float* __restrict__ last,
                                float* __restrict__ out, int step)
{
    int m = blockIdx.x * blockDim.x + threadIdx.x;
    if (m >= NAGENT) return;
    const float* dr = d3 + (size_t)m * (DDIM / 4);
    float v0 = bd4[0], v1 = bd4[1];
    for (int q = 0; q < DDIM / 4; ++q) { v0 += Wd4[q] * dr[q]; v1 += Wd4[DDIM/4 + q] * dr[q]; }
    float p0 = v0 + last[m*4 + 0], p1 = v1 + last[m*4 + 1];
    last[m*4+0] = p0; last[m*4+1] = p1; last[m*4+2] = v0; last[m*4+3] = v1;
    float* o = out + (size_t)m * NNEXT * 4 + (size_t)step * 4;
    o[0] = p0; o[1] = p1; o[2] = v0; o[3] = v1;
}

extern "C" void kernel_launch(void* const* d_in, const int* in_sizes, int n_in,
                              void* d_out, int out_size, void* d_ws, size_t ws_size,
                              hipStream_t stream)
{
    (void)in_sizes; (void)n_in; (void)out_size; (void)ws_size;
    const float* obsv = (const float*)d_in[0];
    const float* noise= (const float*)d_in[1];
    const float* Wemb = (const float*)d_in[2];
    const float* bemb = (const float*)d_in[3];
    const float* Wih  = (const float*)d_in[4];
    const float* Whh  = (const float*)d_in[5];
    const float* bih  = (const float*)d_in[6];
    const float* bhh  = (const float*)d_in[7];
    const float* Wf1  = (const float*)d_in[8];
    const float* bf1  = (const float*)d_in[9];
    const float* Wf2  = (const float*)d_in[10];
    const float* bf2  = (const float*)d_in[11];
    const float* Wf3  = (const float*)d_in[12];
    const float* bf3  = (const float*)d_in[13];
    const float* Wa   = (const float*)d_in[14];
    const float* ba   = (const float*)d_in[15];
    const float* Wd1  = (const float*)d_in[16];
    const float* bd1  = (const float*)d_in[17];
    const float* Wd2  = (const float*)d_in[18];
    const float* bd2  = (const float*)d_in[19];
    const float* Wd3  = (const float*)d_in[20];
    const float* bd3  = (const float*)d_in[21];
    const float* Wd4  = (const float*)d_in[22];
    const float* bd4  = (const float*)d_in[23];
    float* out = (float*)d_out;

    // workspace carve-out (floats); total ~2.4M floats (~9.6 MB)
    float* ws    = (float*)d_ws;
    float* x_emb = ws;  ws += NAGENT * TOBS * HDIM;   // 786432
    float* xe    = ws;  ws += NAGENT * HDIM;          //  98304
    float* h     = ws;  ws += NAGENT * HDIM;
    float* c     = ws;  ws += NAGENT * HDIM;
    float* z     = ws;  ws += NAGENT * 4 * HDIM;      // 393216
    float* Whb   = ws;  ws += NAGENT * HDIM;          // h @ Wa^T + ba
    float* sigma = ws;  ws += NGROUP * GSIZE * GSIZE; //  49152
    float* wsum  = ws;  ws += NAGENT * HDIM;          // weighted
    float* u0    = ws;  ws += NAGENT * DDIM;          // 245760
    float* d1    = ws;  ws += NAGENT * DDIM;
    float* d2    = ws;  ws += NAGENT * (DDIM / 2);
    float* d3    = ws;  ws += NAGENT * (DDIM / 4);
    float* last  = ws;  ws += NAGENT * 4;

    const int EB = (NAGENT * HDIM + 255) / 256;   // 768*128 threads

    init_kernel<<<EB, 256, 0, stream>>>(obsv, h, c, last);
    embed_kernel<<<(NAGENT * TOBS * HDIM + 255) / 256, 256, 0, stream>>>(
        obsv, Wemb, bemb, x_emb, NAGENT * TOBS);

    // ---- encoder: 8 dependent LSTM steps -------------------------------
    for (int t = 0; t < TOBS; ++t) {
        GEMM(HDIM, TOBS * HDIM, HDIM, HDIM,
             x_emb + t * HDIM, Wih, h, Whh, bih, bhh,
             z, 4 * HDIM, 4 * HDIM, 0);
        lstm_gates_kernel<<<EB, 256, 0, stream>>>(z, h, c);
    }

    // ---- social attention (computed once, on encoder-final h) ----------
    GEMM(HDIM, HDIM, 0, 0, h, Wa, nullptr, nullptr, ba, nullptr,
         Whb, HDIM, HDIM, 0);
    social_sigma_kernel<<<(NGROUP * GSIZE * GSIZE + 255) / 256, 256, 0, stream>>>(
        obsv, Wf1, bf1, Wf2, bf2, Wf3, bf3, Whb, sigma);
    softmax_rows_kernel<<<(NGROUP * GSIZE * 32 + 255) / 256, 256, 0, stream>>>(sigma);
    attn_apply_kernel<<<EB, 256, 0, stream>>>(sigma, h, wsum);
    u0_static_kernel<<<(NAGENT * (FDIM + NZDIM) + 255) / 256, 256, 0, stream>>>(
        wsum, noise, u0);

    // ---- decoder: 12 dependent steps -----------------------------------
    for (int s = 0; s < NNEXT; ++s) {
        u0_h_kernel<<<EB, 256, 0, stream>>>(h, u0);
        GEMM(DDIM, DDIM, 0, 0, u0, Wd1, nullptr, nullptr, bd1, nullptr,
             d1, DDIM, DDIM, 2);
        GEMM(DDIM, DDIM, 0, 0, d1, Wd2, nullptr, nullptr, bd2, nullptr,
             d2, DDIM / 2, DDIM / 2, 2);
        GEMM(DDIM / 2, DDIM / 2, 0, 0, d2, Wd3, nullptr, nullptr, bd3, nullptr,
             d3, DDIM / 4, DDIM / 4, 0);
        dec_tail_kernel<<<(NAGENT + 255) / 256, 256, 0, stream>>>(
            d3, Wd4, bd4, last, out, s);
        embed_kernel<<<EB, 256, 0, stream>>>(last, Wemb, bemb, xe, NAGENT);
        GEMM(HDIM, HDIM, HDIM, HDIM, xe, Wih, h, Whh, bih, bhh,
             z, 4 * HDIM, 4 * HDIM, 0);
        lstm_gates_kernel<<<EB, 256, 0, stream>>>(z, h, c);
    }
}
